// LiquidCell3D_34935263985783
// MI455X (gfx1250) — matrix operand order repaired and checked by reading, not compiled, not run on hardware
//
#include <hip/hip_runtime.h>

typedef __attribute__((ext_vector_type(16))) _Float16 v16h;
typedef __attribute__((ext_vector_type(8)))  _Float16 v8h;
typedef __attribute__((ext_vector_type(4)))  _Float16 v4h;
typedef __attribute__((ext_vector_type(8)))  float    v8f;
typedef __attribute__((ext_vector_type(4)))  float    v4f;

#define LC_WAVES     4      // waves per block (128 threads)
#define F32_PITCH    68     // padded f32 LDS row pitch (dwords) -> conflict-free
#define F16_PITCH    72     // padded f16 LDS row pitch (halves), 144B, 16B aligned

__global__ __launch_bounds__(128)
void liquid_cell_kernel(const float* __restrict__ hIn,
                        const float* __restrict__ uIn,
                        const float* __restrict__ W_h,
                        const float* __restrict__ U_h,
                        const float* __restrict__ b_h,
                        const float* __restrict__ W_tau,
                        const float* __restrict__ U_tau,
                        const float* __restrict__ b_tau,
                        const float* __restrict__ W_out,
                        const float* __restrict__ b_out,
                        float* __restrict__ hNewOut,
                        float* __restrict__ vOut,
                        int numTokens)
{
    // Block-shared f16 copies of the two 64x64 weight matrices (B operands).
    __shared__ _Float16 sW16[2][64 * F16_PITCH];
    // Per-wave tiles.
    __shared__ float    sH  [LC_WAVES][16 * F32_PITCH];   // f32 h tile
    __shared__ float    sHN [LC_WAVES][16 * F32_PITCH];   // f32 h_new tile
    __shared__ _Float16 sH16[LC_WAVES][16 * F16_PITCH];   // f16 h tile (A fragments)
    __shared__ float    sU  [LC_WAVES][64];               // u tile (16 x 3, padded 4)

    const int lane = threadIdx.x & 31;
    const int wv   = threadIdx.x >> 5;
    const int lh   = lane >> 4;       // 0 = low half, 1 = high half of wave
    const int ll   = lane & 15;

    float*    Hs  = sH[wv];
    float*    HNs = sHN[wv];
    _Float16* H16 = sH16[wv];
    float*    Us  = sU[wv];

    // ---- stage f16 weights into block-shared LDS (cooperative, once) ----
    // 64x64 floats per matrix; 128 threads x 4 floats x 8 iters.
    {
        const float* srcs[2] = { W_h, W_tau };
#pragma unroll
        for (int mmat = 0; mmat < 2; ++mmat) {
            const float* src = srcs[mmat];
            _Float16*    dst = sW16[mmat];
#pragma unroll
            for (int i = 0; i < 8; ++i) {
                const int e   = (i * 128 + (int)threadIdx.x) * 4;  // element index
                const int row = e >> 6;
                const int col = e & 63;
                v4f x = *(const v4f*)(src + e);
                v4h xh;
#pragma unroll
                for (int j = 0; j < 4; ++j) xh[j] = (_Float16)x[j];
                *(v4h*)&dst[row * F16_PITCH + col] = xh;
            }
        }
    }
    __syncthreads();

    const int tiles      = numTokens >> 4;
    const int waveGlobal = blockIdx.x * LC_WAVES + wv;
    const int nWaves     = gridDim.x * LC_WAVES;

    for (int tile = waveGlobal; tile < tiles; tile += nWaves) {
        const int tok0 = tile << 4;

        // Prefetch next tile's h rows into cache (global_prefetch_b8).
        {
            const int nextTile = tile + nWaves;
            if (nextTile < tiles) {
                const char* p = (const char*)(hIn + (size_t)(nextTile << 4) * 64);
                __builtin_prefetch(p + lane * 128, 0, 1);   // 32 lanes x 128B = 4KB tile
            }
        }

        // ---- phase 1: stage 16x64 h tile (f32 + f16) and u tile into LDS ----
        {
            const v4f* src = (const v4f*)(hIn + (size_t)tok0 * 64);
#pragma unroll
            for (int i = 0; i < 8; ++i) {
                v4f x = src[i * 32 + lane];                 // coalesced b128
                const int row = i * 2 + lh;
                const int col = ll * 4;
                *(v4f*)&Hs[row * F32_PITCH + col] = x;
                v4h xh;
#pragma unroll
                for (int j = 0; j < 4; ++j) xh[j] = (_Float16)x[j];
                *(v4h*)&H16[row * F16_PITCH + col] = xh;
            }
            if (lane < 16) {
                const float* us = uIn + (size_t)(tok0 + lane) * 3;
                Us[lane * 4 + 0] = us[0];
                Us[lane * 4 + 1] = us[1];
                Us[lane * 4 + 2] = us[2];
            }
        }
        asm volatile("s_wait_dscnt 0" ::: "memory");  // LDS tile visible to all lanes

        // ---- phase 2: D = A x B, K=64 in two 32-deep chunks ----
        // A (16-bit 16x32): lane holds row (l&15); lanes<16 K{0-7,16-23},
        //                   lanes>=16 K{8-15,24-31}.
        // B (16-bit 32x16): lane holds column n=(l&15); same K split as A.
        v8f accW[4] = {}, accT[4] = {};
#pragma unroll
        for (int kc = 0; kc < 2; ++kc) {
            v16h a;
            {
                const int rowb = ll * F16_PITCH;
                const int koff = kc * 32 + lh * 8;
                v8h lo = *(const v8h*)&H16[rowb + koff];
                v8h hi = *(const v8h*)&H16[rowb + koff + 16];
#pragma unroll
                for (int j = 0; j < 8; ++j) { a[j] = lo[j]; a[8 + j] = hi[j]; }
            }
            const int kb = kc * 32 + lh * 8;
#pragma unroll
            for (int t = 0; t < 4; ++t) {
                const int rb = (t * 16 + ll) * F16_PITCH + kb;
                v16h bw, bt;
                {
                    v8h lo = *(const v8h*)&sW16[0][rb];
                    v8h hi = *(const v8h*)&sW16[0][rb + 16];
#pragma unroll
                    for (int j = 0; j < 8; ++j) { bw[j] = lo[j]; bw[8 + j] = hi[j]; }
                    lo = *(const v8h*)&sW16[1][rb];
                    hi = *(const v8h*)&sW16[1][rb + 16];
#pragma unroll
                    for (int j = 0; j < 8; ++j) { bt[j] = lo[j]; bt[8 + j] = hi[j]; }
                }
                accW[t] = __builtin_amdgcn_wmma_f32_16x16x32_f16(
                    false, a, false, bw, (short)0, accW[t], false, false);
                accT[t] = __builtin_amdgcn_wmma_f32_16x16x32_f16(
                    false, a, false, bt, (short)0, accT[t], false, false);
            }
        }

        // ---- phase 3: elementwise liquid-cell update in C/D layout ----
        // C/D: VGPR r -> M = r + 8*(lane>=16); lane&15 -> N
#pragma unroll
        for (int t = 0; t < 4; ++t) {
            const int n   = t * 16 + ll;
            const float bhv = b_h[n];
            const float btv = b_tau[n];
            const float uh0 = U_h[n * 3 + 0], uh1 = U_h[n * 3 + 1], uh2 = U_h[n * 3 + 2];
            const float ut0 = U_tau[n * 3 + 0], ut1 = U_tau[n * 3 + 1], ut2 = U_tau[n * 3 + 2];
#pragma unroll
            for (int r = 0; r < 8; ++r) {
                const int m  = r + lh * 8;
                const float u0 = Us[m * 4 + 0];
                const float u1 = Us[m * 4 + 1];
                const float u2 = Us[m * 4 + 2];
                float raw = accT[t][r] + u0 * ut0 + u1 * ut1 + u2 * ut2 + btv;
                float pre = accW[t][r] + u0 * uh0 + u1 * uh1 + u2 * uh2 + bhv;
                // stable softplus: max(x,0) + log(1 + exp(-|x|))
                float sp  = fmaxf(raw, 0.0f) + __logf(1.0f + __expf(-fabsf(raw)));
                float tau = 0.1f + 9.9f * sp;
                float fh  = __builtin_amdgcn_rcpf(1.0f + __expf(-pre));   // sigmoid
                float hv  = Hs[m * F32_PITCH + n];
                float hn  = hv + 0.1f * (fh - hv * __builtin_amdgcn_rcpf(tau));
                HNs[m * F32_PITCH + n] = hn;
                hNewOut[(size_t)(tok0 + m) * 64 + n] = hn;
            }
        }
        asm volatile("s_wait_dscnt 0" ::: "memory");  // h_new tile visible

        // ---- phase 4: v = tanh(h_new @ W_out^T + b_out), 3 outs x 16 tokens ----
        // lanes 0-15 handle o = 0,2 for token (l&15); lanes 16-31 handle o = 1
        const int tk = ll;
#pragma unroll 1
        for (int o = lh; o < 3; o += 2) {
            float s = b_out[o];
            const float* wo = W_out + o * 64;
#pragma unroll
            for (int f = 0; f < 64; f += 4) {
                s += HNs[tk * F32_PITCH + f + 0] * wo[f + 0]
                   + HNs[tk * F32_PITCH + f + 1] * wo[f + 1]
                   + HNs[tk * F32_PITCH + f + 2] * wo[f + 2]
                   + HNs[tk * F32_PITCH + f + 3] * wo[f + 3];
            }
            float x = fminf(fmaxf(s, -15.0f), 15.0f);
            float e = __expf(2.0f * x);
            vOut[(size_t)(tok0 + tk) * 3 + o] =
                (e - 1.0f) * __builtin_amdgcn_rcpf(e + 1.0f);
        }
    }
}

extern "C" void kernel_launch(void* const* d_in, const int* in_sizes, int n_in,
                              void* d_out, int out_size, void* d_ws, size_t ws_size,
                              hipStream_t stream) {
    const float* h     = (const float*)d_in[0];
    const float* u     = (const float*)d_in[1];
    const float* W_h   = (const float*)d_in[2];
    const float* U_h   = (const float*)d_in[3];
    const float* b_h   = (const float*)d_in[4];
    const float* W_tau = (const float*)d_in[5];
    const float* U_tau = (const float*)d_in[6];
    const float* b_tau = (const float*)d_in[7];
    const float* W_out = (const float*)d_in[8];
    const float* b_out = (const float*)d_in[9];

    const int numTokens = in_sizes[0] / 64;                  // B*N
    float* hNewOut = (float*)d_out;                          // (B,N,64)
    float* vOut    = (float*)d_out + (size_t)numTokens * 64; // (B,N,3)

    const int tiles        = numTokens / 16;
    const int tilesPerWave = 8;
    int blocks = (tiles + LC_WAVES * tilesPerWave - 1) / (LC_WAVES * tilesPerWave);
    if (blocks < 1) blocks = 1;

    liquid_cell_kernel<<<blocks, 128, 0, stream>>>(
        h, u, W_h, U_h, b_h, W_tau, U_tau, b_tau, W_out, b_out,
        hNewOut, vOut, numTokens);
}
